// ViT_RSNN_NoConv_59957743452581
// MI455X (gfx1250) — matrix operand order repaired and checked by reading, not compiled, not run on hardware
//
#include <hip/hip_runtime.h>
#include <hip/hip_bf16.h>
#include <math.h>

typedef __attribute__((ext_vector_type(16))) _Float16 v16h;
typedef __attribute__((ext_vector_type(8)))  float    v8f;

#define NH 1000
#define NI 46656
#define NB 32

// Low 32 bits of a generic pointer to __shared__ == wave-relative LDS byte
// address (flat LDS aperture keeps the offset in addr[31:0], per ISA 10.2).
__device__ __forceinline__ unsigned lds_addr32(const void* p)
{
    return (unsigned)(size_t)p;
}

// ---------------------------------------------------------------------------
// Generic WMMA GEMM: out = A[32,K] @ W[K,N] + bias[N]
//   mode 0: store cur
//   mode 1: leaky epilogue: reset=(mem>thr); mem=beta*mem+cur-reset*thr;
//           spk=(mem>thr); store mem & spk
// Block = 64 threads = 2 waves; wave w handles rows w*16..w*16+15; one 16-col
// N tile per block. B tile (32x16 f32) staged to LDS with async global->LDS
// copies in the fast path; f32->f16 conversion happens on the LDS read that
// builds the WMMA B fragment.
// ---------------------------------------------------------------------------
__global__ __launch_bounds__(64)
void gemm_wmma_kernel(const float* __restrict__ A, const float* __restrict__ W,
                      const float* __restrict__ bias, int K, int N,
                      int mode, float beta, float thr,
                      float* __restrict__ outv, float* __restrict__ mem)
{
    __shared__ __align__(16) float lbF[32 * 16];
    const int tid   = threadIdx.x;
    const int wave  = tid >> 5;
    const int l     = tid & 31;
    const int r     = l & 15;     // row-in-tile (A) / col-in-tile (B,C)
    const int half  = l >> 4;     // lane group
    const int n0    = blockIdx.x * 16;
    const int mbase = wave * 16;
    const int row   = mbase + r;
    const bool colfull = (n0 + 16 <= N);

    // This thread's staging slot: 8 consecutive floats of the B tile.
    const int skk = tid >> 1;          // k within chunk: 0..31
    const int scc = (tid & 1) * 8;     // n within tile : 0 or 8
    float* myslot = &lbF[skk * 16 + scc];
    const unsigned myslot_lds = lds_addr32(myslot);

    v8f c = {};
    for (int k0 = 0; k0 < K; k0 += 32) {
        const bool kfull = (k0 + 32 <= K);

        // ---- stage B tile [32 k][16 n] into LDS (f32) ----
        if (kfull && colfull) {
            const float* gp = W + (size_t)(k0 + skk) * N + (n0 + scc);
            asm volatile(
                "global_load_async_to_lds_b128 %0, %1, off\n\t"
                "global_load_async_to_lds_b128 %0, %1, off offset:16"
                :: "v"(myslot_lds), "v"(gp) : "memory");
            asm volatile("s_wait_asynccnt 0" ::: "memory");
        } else {
            #pragma unroll
            for (int u = 0; u < 8; ++u) {
                int kg = k0 + skk;
                int n  = n0 + scc + u;
                myslot[u] = (kg < K && n < N) ? W[(size_t)kg * N + n] : 0.0f;
            }
        }
        __syncthreads();

        // ---- A fragment (16-bit A 16x32 layout): two contiguous 8-float runs
        v16h a, b;
        if (kfull) {
            if (k0 + 64 <= K)
                __builtin_prefetch(A + (size_t)row * K + k0 + 32 + half * 8, 0, 1);
            const float4* pa = (const float4*)(A + (size_t)row * K + k0 + half * 8);
            float4 a0 = pa[0], a1 = pa[1];   // K = k0+half*8 + 0..7
            float4 a2 = pa[4], a3 = pa[5];   // K = k0+half*8 + 16..23
            a[0]  = (_Float16)a0.x; a[1]  = (_Float16)a0.y;
            a[2]  = (_Float16)a0.z; a[3]  = (_Float16)a0.w;
            a[4]  = (_Float16)a1.x; a[5]  = (_Float16)a1.y;
            a[6]  = (_Float16)a1.z; a[7]  = (_Float16)a1.w;
            a[8]  = (_Float16)a2.x; a[9]  = (_Float16)a2.y;
            a[10] = (_Float16)a2.z; a[11] = (_Float16)a2.w;
            a[12] = (_Float16)a3.x; a[13] = (_Float16)a3.y;
            a[14] = (_Float16)a3.z; a[15] = (_Float16)a3.w;
        } else {
            #pragma unroll
            for (int e = 0; e < 16; ++e) {
                int k = k0 + half * 8 + e + ((e >= 8) ? 8 : 0);
                a[e] = (k < K) ? (_Float16)A[(size_t)row * K + k] : (_Float16)0.0f;
            }
        }
        // ---- B fragment: same K order as A: K = k0+half*8 + 0..7, 16..23
        #pragma unroll
        for (int e = 0; e < 16; ++e)
            b[e] = (_Float16)lbF[(half * 8 + e + ((e >= 8) ? 8 : 0)) * 16 + r];

        c = __builtin_amdgcn_wmma_f32_16x16x32_f16(false, a, false, b,
                                                   (short)0, c, false, false);
        __syncthreads();
    }

    // C layout: VGPR j -> M = mbase + half*8 + j, N = n0 + r
    #pragma unroll
    for (int j = 0; j < 8; ++j) {
        int m = mbase + half * 8 + j;
        int n = n0 + r;
        if (n < N) {
            float cur = c[j] + bias[n];
            size_t idx = (size_t)m * N + n;
            if (mode == 0) {
                outv[idx] = cur;
            } else {
                float mm = mem[idx];
                float reset = (mm > thr) ? 1.0f : 0.0f;
                mm = beta * mm + cur - reset * thr;
                mem[idx]  = mm;
                outv[idx] = (mm > thr) ? 1.0f : 0.0f;
            }
        }
    }
}

// ---------------------------------------------------------------------------
// Patch embedding: x_flat[b, p*64+e] = patch(b,p) . W_pe[:,e] + b_pe[e] + pos[p,e]
// blockIdx.x = b*729 + p ; 64 threads, one embed dim each; patch in LDS.
// ---------------------------------------------------------------------------
__global__ __launch_bounds__(64)
void patch_embed_kernel(const float* __restrict__ img, const float* __restrict__ W_pe,
                        const float* __restrict__ b_pe, const float* __restrict__ pos,
                        float* __restrict__ x_flat)
{
    __shared__ float patch[256];
    int bp = blockIdx.x;
    int b = bp / 729, p = bp % 729;
    int ph = p / 27, pw = p % 27;
    const float* im = img + (size_t)b * 224 * 224;
    for (int i = threadIdx.x; i < 256; i += 64) {
        int ii = i >> 4, jj = i & 15;
        patch[i] = im[(ph * 8 + ii) * 224 + (pw * 8 + jj)];
    }
    __syncthreads();
    int e = threadIdx.x;
    float acc = b_pe[e] + pos[p * 64 + e];
    for (int k = 0; k < 256; ++k) acc += patch[k] * W_pe[k * 64 + e];
    x_flat[(size_t)b * NI + p * 64 + e] = acc;
}

// ---------------------------------------------------------------------------
// Leaky1: two consecutive leaky updates of mem1 with the same cur1
// ---------------------------------------------------------------------------
__global__ void leaky1_kernel(const float* __restrict__ cur1, float* __restrict__ mem1,
                              float* __restrict__ spk1, float* __restrict__ spk1b, int n)
{
    int i = blockIdx.x * blockDim.x + threadIdx.x;
    if (i >= n) return;
    float cur = cur1[i], m = mem1[i];
    float reset = (m > 0.5f) ? 1.0f : 0.0f;
    m = 0.8f * m + cur - reset * 0.5f;
    float s1 = (m > 0.5f) ? 1.0f : 0.0f;
    m = 0.8f * m + cur - s1 * 0.5f;            // reset of 2nd update == s1
    float s1b = (m > 0.5f) ? 1.0f : 0.0f;
    mem1[i] = m; spk1[i] = s1; spk1b[i] = s1b;
}

// ---------------------------------------------------------------------------
// RLeaky: mem_r = 0.95*mem_r + spk1@WR + bR + spk_r_prev@Vw + Vb - reset
// ---------------------------------------------------------------------------
__global__ void rleaky_kernel(const float* __restrict__ spk1, const float* __restrict__ WR,
                              const float* __restrict__ bR, const float* __restrict__ srP,
                              const float* __restrict__ Vw, const float* __restrict__ Vb,
                              float* __restrict__ mem_r, float* __restrict__ srN)
{
    int i = blockIdx.x * blockDim.x + threadIdx.x;
    if (i >= NB * 100) return;
    int b = i / 100, j = i % 100;
    const float* s1 = spk1 + (size_t)b * NH;
    float acc = bR[j];
    for (int h = 0; h < NH; ++h) acc += s1[h] * WR[h * 100 + j];
    const float* sr = srP + b * 100;
    float rec = Vb[j];
    for (int h = 0; h < 100; ++h) rec += sr[h] * Vw[h * 100 + j];
    float m = mem_r[i];
    float reset = (m > 1.0f) ? 1.0f : 0.0f;
    m = 0.95f * m + acc + rec - reset;
    mem_r[i] = m;
    srN[i] = (m > 1.0f) ? 1.0f : 0.0f;
}

// ---------------------------------------------------------------------------
// Attention + output projection + output leaky, fused. One block per batch.
// att_i = sum_j softmax_j(q_i*k_j/sqrt(1000)) * v_j ; out = att.Wo + bo
// ---------------------------------------------------------------------------
__global__ __launch_bounds__(256)
void attn_out_kernel(const float* __restrict__ q, const float* __restrict__ kmat,
                     const float* __restrict__ vmat, const float* __restrict__ Wo,
                     const float* __restrict__ bo, float* __restrict__ mem_out,
                     float* __restrict__ spk_out)
{
    __shared__ float kk[NH], vv[NH], att[NH];
    __shared__ float red[256];
    int b = blockIdx.x;
    for (int i = threadIdx.x; i < NH; i += 256) {
        kk[i] = kmat[(size_t)b * NH + i];
        vv[i] = vmat[(size_t)b * NH + i];
    }
    __syncthreads();
    const float inv_s = 0.031622776601683794f;   // 1/sqrt(1000)
    for (int i = threadIdx.x; i < NH; i += 256) {
        float qi = q[(size_t)b * NH + i] * inv_s;
        float mx = -3.4e38f;
        for (int j = 0; j < NH; ++j) { float t = qi * kk[j]; mx = fmaxf(mx, t); }
        float se = 0.0f, sv = 0.0f;
        for (int j = 0; j < NH; ++j) {
            float e = __expf(qi * kk[j] - mx);
            se += e; sv += e * vv[j];
        }
        att[i] = sv / se;
    }
    __syncthreads();
    float part = 0.0f;
    for (int i = threadIdx.x; i < NH; i += 256) part += att[i] * Wo[i];
    red[threadIdx.x] = part;
    __syncthreads();
    for (int s = 128; s > 0; s >>= 1) {
        if (threadIdx.x < s) red[threadIdx.x] += red[threadIdx.x + s];
        __syncthreads();
    }
    if (threadIdx.x == 0) {
        float cur = red[0] + bo[0];
        float m = mem_out[b];
        float reset = (m > 0.5f) ? 1.0f : 0.0f;
        m = 0.8f * m + cur - reset * 0.5f;
        mem_out[b] = m;
        spk_out[b] = (m > 0.5f) ? 1.0f : 0.0f;
    }
}

// ---------------------------------------------------------------------------
// Per-step scalar bookkeeping: total/prev/reg + out accumulation
// scalars: [0]=total  [1]=prev  [2]=reg
// ---------------------------------------------------------------------------
__global__ __launch_bounds__(256)
void finalize_kernel(const float* __restrict__ spk1, const float* __restrict__ spk3,
                     const float* __restrict__ spk_r, const float* __restrict__ spk_out,
                     float* __restrict__ scalars, float* __restrict__ out_acc, int t)
{
    __shared__ float red[256];
    float p = 0.0f;
    for (int i = threadIdx.x; i < NB * NH; i += 256)
        p += (spk1[i] + spk3[i]) * (1.0f / (NB * NH));
    for (int i = threadIdx.x; i < NB * 100; i += 256)
        p += spk_r[i] * (1.0f / (NB * 100));
    for (int i = threadIdx.x; i < NB; i += 256)
        p += spk_out[i] * (1.0f / NB);
    red[threadIdx.x] = p;
    __syncthreads();
    for (int s = 128; s > 0; s >>= 1) {
        if (threadIdx.x < s) red[threadIdx.x] += red[threadIdx.x + s];
        __syncthreads();
    }
    if (threadIdx.x < NB) out_acc[threadIdx.x] += spk_out[threadIdx.x];
    if (threadIdx.x == 0) {
        float total = scalars[0] + red[0];
        float prev  = scalars[1];
        float reg   = scalars[2] + 0.0058f * total
                    + ((t > 0) ? 0.0058f * fabsf(total - prev) : 0.0f);
        scalars[0] = total; scalars[1] = total; scalars[2] = reg;
    }
}

__global__ void zero_kernel(float* __restrict__ p, int n)
{
    int i = blockIdx.x * blockDim.x + threadIdx.x;
    if (i < n) p[i] = 0.0f;
}

__global__ void writeout_kernel(const float* __restrict__ out_acc,
                                const float* __restrict__ scalars,
                                float* __restrict__ out)
{
    int i = threadIdx.x;
    if (i < NB) out[i] = out_acc[i] * 0.2f;          // sum over 5 steps / 5
    if (i == NB) out[NB] = scalars[2] * 0.2f;        // reg / 5
}

// ---------------------------------------------------------------------------
extern "C" void kernel_launch(void* const* d_in, const int* in_sizes, int n_in,
                              void* d_out, int out_size, void* d_ws, size_t ws_size,
                              hipStream_t stream)
{
    (void)in_sizes; (void)n_in; (void)out_size; (void)ws_size;
    const float* data = (const float*)d_in[0];
    const float* W_pe = (const float*)d_in[1];
    const float* b_pe = (const float*)d_in[2];
    const float* pos  = (const float*)d_in[3];
    const float* W1 = (const float*)d_in[4];  const float* b1 = (const float*)d_in[5];
    const float* W2 = (const float*)d_in[6];  const float* b2 = (const float*)d_in[7];
    const float* W3 = (const float*)d_in[8];  const float* b3 = (const float*)d_in[9];
    const float* WR = (const float*)d_in[10]; const float* bR = (const float*)d_in[11];
    const float* Vw = (const float*)d_in[12]; const float* Vb = (const float*)d_in[13];
    const float* Wq = (const float*)d_in[14]; const float* bq = (const float*)d_in[15];
    const float* Wk = (const float*)d_in[16]; const float* bk = (const float*)d_in[17];
    const float* Wv = (const float*)d_in[18]; const float* bv = (const float*)d_in[19];
    const float* Wo = (const float*)d_in[20]; const float* bo = (const float*)d_in[21];
    float* out = (float*)d_out;

    // workspace layout (floats); every offset is a multiple of 4 floats (16 B)
    float* w = (float*)d_ws;
    size_t o = 0;
    float* x_flat = w + o; o += (size_t)NB * NI;   // 1,492,992
    float* cur1   = w + o; o += NB * NH;
    float* spk1   = w + o; o += NB * NH;
    float* spk1b  = w + o; o += NB * NH;
    float* spk2   = w + o; o += NB * NH;
    float* spk3   = w + o; o += NB * NH;
    float* qbuf   = w + o; o += NB * NH;
    float* kbuf   = w + o; o += NB * NH;
    float* vbuf   = w + o; o += NB * NH;
    // --- zero-initialized state block ---
    size_t state_begin = o;
    float* mem1    = w + o; o += NB * NH;
    float* mem2    = w + o; o += NB * NH;
    float* mem3    = w + o; o += NB * NH;
    float* mem_r   = w + o; o += NB * 100;
    float* srA     = w + o; o += NB * 100;
    float* srB     = w + o; o += NB * 100;
    float* mem_out = w + o; o += NB;
    float* spk_out = w + o; o += NB;
    float* out_acc = w + o; o += NB;
    float* scalars = w + o; o += 4;
    int state_len = (int)(o - state_begin);

    zero_kernel<<<(state_len + 255) / 256, 256, 0, stream>>>(w + state_begin, state_len);

    // patch embedding -> x_flat [32, 46656]
    patch_embed_kernel<<<NB * 729, 64, 0, stream>>>(data, W_pe, b_pe, pos, x_flat);

    // cur1 = x_flat @ W1 + b1   (constant across steps)
    dim3 ggrid((NH + 15) / 16);
    gemm_wmma_kernel<<<ggrid, 64, 0, stream>>>(x_flat, W1, b1, NI, NH,
                                               0, 0.0f, 0.0f, cur1, nullptr);

    float* srP = srA;   // spk_r double buffers
    float* srN = srB;
    for (int t = 0; t < 5; ++t) {
        leaky1_kernel<<<(NB * NH + 255) / 256, 256, 0, stream>>>(cur1, mem1, spk1, spk1b, NB * NH);
        gemm_wmma_kernel<<<ggrid, 64, 0, stream>>>(spk1b, W2, b2, NH, NH,
                                                   1, 0.8f, 0.5f, spk2, mem2);
        gemm_wmma_kernel<<<ggrid, 64, 0, stream>>>(spk2, W3, b3, NH, NH,
                                                   1, 0.8f, 0.5f, spk3, mem3);
        rleaky_kernel<<<(NB * 100 + 127) / 128, 128, 0, stream>>>(spk1, WR, bR, srP,
                                                                  Vw, Vb, mem_r, srN);
        gemm_wmma_kernel<<<ggrid, 64, 0, stream>>>(srN, Wq, bq, 100, NH,
                                                   0, 0.0f, 0.0f, qbuf, nullptr);
        gemm_wmma_kernel<<<ggrid, 64, 0, stream>>>(spk3, Wk, bk, NH, NH,
                                                   0, 0.0f, 0.0f, kbuf, nullptr);
        gemm_wmma_kernel<<<ggrid, 64, 0, stream>>>(spk3, Wv, bv, NH, NH,
                                                   0, 0.0f, 0.0f, vbuf, nullptr);
        attn_out_kernel<<<NB, 256, 0, stream>>>(qbuf, kbuf, vbuf, Wo, bo, mem_out, spk_out);
        finalize_kernel<<<1, 256, 0, stream>>>(spk1, spk3, srN, spk_out, scalars, out_acc, t);
        float* tmp = srP; srP = srN; srN = tmp;
    }

    writeout_kernel<<<1, 64, 0, stream>>>(out_acc, scalars, out);
}
